// OCN_20564303413655
// MI455X (gfx1250) — matrix operand order repaired and checked by reading, not compiled, not run on hardware
//
#include <hip/hip_runtime.h>
#include <hip/hip_bf16.h>
#include <math.h>

// ---------------------------------------------------------------------------
// Problem constants (reference: H=1024, S=512, B=16, L=4).
// fp in [300,450) -> Lo <= 212 (pad to 256), Ld <= 449 (pad to 512).
// Padded positions are masked, matching the reference's mask semantics.
// ---------------------------------------------------------------------------
#define H_    1024
#define S_    512
#define BL_   64      // B*L
#define BSZ_  16
#define NL_   4
#define LO_   256
#define LD_   512
#define NEGV  (-1e30f)

typedef _Float16 h16;
typedef __attribute__((ext_vector_type(16))) _Float16 v16h;
typedef __attribute__((ext_vector_type(8)))  _Float16 v8h;
typedef __attribute__((ext_vector_type(8)))  float    v8f;

// ---------------------------------------------------------------------------
// Batched NT GEMM on WMMA f16->f32:  C[b] (MxN, f32) = A[b](MxK) * B[b](NxK)^T
// A, B row-major f16.  M mult of 32, N mult of 64, K mult of 32.
// One wave per 32x64 C tile (2 A-frags reused across 4 B-frags -> 8 WMMA/k-step).
// Fragment layouts used here (wave32; A and B share contiguous k order, not the ISA table's):
//   A 16x32: lane L -> m = L&15, g = L>>4; elem e -> k = e + 16g
//            => two contiguous 8-half (16B) loads per lane.
//   B 32x16: lane L -> n = L&15, g = L>>4; elem e -> k = e + 16g
//            => one contiguous 16-half (32B) load per lane from row-major NxK.
//   C 16x16: lane L -> n = L&15; vgpr r -> m = r + 8*(L>>4)
// ---------------------------------------------------------------------------
__global__ void __launch_bounds__(256)
wmma_gemm_nt(const h16* __restrict__ A, long long sA, int lda,
             const h16* __restrict__ B, long long sB, int ldb,
             float* __restrict__ C, long long sC, int ldc,
             int M, int N, int K, int nbatch, int accumulate)
{
    const int lane = threadIdx.x;
    const int wid  = threadIdx.y;
    const int tm = M >> 5, tn = N >> 6;
    long long tile  = (long long)blockIdx.x * 8 + wid;
    long long total = (long long)nbatch * tm * tn;
    if (tile >= total) return;                 // whole-wave uniform: EXEC all 1s below
    int b  = (int)(tile / (tm * tn));
    int t  = (int)(tile % (tm * tn));
    int m0 = (t / tn) << 5;
    int n0 = (t % tn) << 6;
    const h16* Ab = A + (long long)b * sA;
    const h16* Bb = B + (long long)b * sB;
    float*     Cb = C + (long long)b * sC;
    const int r16 = lane & 15;
    const int g   = lane >> 4;

    v8f acc[2][4];
    if (accumulate) {
        for (int i = 0; i < 2; ++i)
            for (int j = 0; j < 4; ++j)
                for (int r = 0; r < 8; ++r)
                    acc[i][j][r] = Cb[(long long)(m0 + 16*i + r + 8*g) * ldc + (n0 + 16*j + r16)];
    } else {
        for (int i = 0; i < 2; ++i)
            for (int j = 0; j < 4; ++j)
                for (int r = 0; r < 8; ++r)
                    acc[i][j][r] = 0.0f;
    }

    for (int k0 = 0; k0 < K; k0 += 32) {
        v16h af[2], bf[4];
        for (int i = 0; i < 2; ++i) {
            const h16* p = Ab + (long long)(m0 + 16*i + r16) * lda + k0 + 16*g;
            v8h lo = *(const v8h*)p;          // elems 0..7  : k = 16g + 0..7
            v8h hi = *(const v8h*)(p + 8);    // elems 8..15 : k = 16g + 8..15
            af[i] = __builtin_shufflevector(lo, hi, 0,1,2,3,4,5,6,7,8,9,10,11,12,13,14,15);
        }
        for (int j = 0; j < 4; ++j) {
            const h16* p = Bb + (long long)(n0 + 16*j + r16) * ldb + k0 + 16*g;
            bf[j] = *(const v16h*)p;          // elems 0..15 : k = 16g + 0..15
        }
        for (int i = 0; i < 2; ++i)
            for (int j = 0; j < 4; ++j)
                acc[i][j] = __builtin_amdgcn_wmma_f32_16x16x32_f16(
                    false, af[i], false, bf[j], (short)0, acc[i][j], false, false);
    }

    for (int i = 0; i < 2; ++i)
        for (int j = 0; j < 4; ++j)
            for (int r = 0; r < 8; ++r)
                Cb[(long long)(m0 + 16*i + r + 8*g) * ldc + (n0 + 16*j + r16)] = acc[i][j][r];
}

// ---------------------------------------------------------------------------
// Prep / gather kernels
// ---------------------------------------------------------------------------
__global__ void prep_opt(const float* __restrict__ ll, const int* __restrict__ am,
                         const long long* __restrict__ fp, const float* __restrict__ w3,
                         float* __restrict__ opt_f32, h16* __restrict__ opt_h,
                         h16* __restrict__ opt3_h, h16* __restrict__ optT_h,
                         float* __restrict__ opt_mask)
{
    int row = blockIdx.x;             // b*LO + t
    int b = row / LO_, t = row % LO_;
    long long idx = fp[b] + t;
    int vidx = (idx < S_);
    int vmask = vidx && (am[b * S_ + (vidx ? (int)idx : 0)] != 0);
    if (threadIdx.x == 0) opt_mask[row] = vmask ? 1.0f : 0.0f;
    const float* src = ll + ((long long)b * S_ + (vidx ? idx : 0)) * H_;
    for (int h = threadIdx.x; h < H_; h += blockDim.x) {
        float x = vidx ? src[h] : 0.0f;
        long long e = (long long)row * H_ + h;
        opt_f32[e] = x;
        opt_h[e]   = (h16)x;
        opt3_h[e]  = (h16)(x * w3[h]);
        optT_h[((long long)b * H_ + h) * LO_ + t] = (h16)x;
    }
}

__global__ void prep_doc(const float* __restrict__ ll, const int* __restrict__ am,
                         const long long* __restrict__ fp, const float* __restrict__ w3,
                         h16* __restrict__ doc_h, h16* __restrict__ doc3_h,
                         h16* __restrict__ docT_h, float* __restrict__ doc_mask)
{
    int row = blockIdx.x;             // b*LD + t
    int b = row / LD_, t = row % LD_;
    int valid = (t < (int)fp[b]) && (am[b * S_ + t] != 0);
    if (threadIdx.x == 0) doc_mask[row] = valid ? 1.0f : 0.0f;
    const float* src = ll + ((long long)b * S_ + t) * H_;
    for (int h = threadIdx.x; h < H_; h += blockDim.x) {
        float x = valid ? src[h] : 0.0f;
        long long e = (long long)row * H_ + h;
        doc_h[e]  = (h16)x;
        doc3_h[e] = (h16)(x * w3[h]);
        docT_h[((long long)b * H_ + h) * LD_ + t] = (h16)x;
    }
}

// ---------------------------------------------------------------------------
// Row dot products: out[row] = sum_h X[row,h]*w[h]    (one wave per row)
// ---------------------------------------------------------------------------
__global__ void rowdot_f32(const float* __restrict__ X, const float* __restrict__ w,
                           float* __restrict__ out, int rows)
{
    int row = blockIdx.x * blockDim.y + threadIdx.y;
    int lane = threadIdx.x;
    if (row >= rows) return;
    float s = 0.0f;
    for (int h = lane; h < H_; h += 32) s += X[(long long)row * H_ + h] * w[h];
    for (int o = 16; o > 0; o >>= 1) s += __shfl_xor(s, o, 32);
    if (lane == 0) out[row] = s;
}

__global__ void rowdot_f16(const h16* __restrict__ X, const float* __restrict__ w,
                           float* __restrict__ out, int rows)
{
    int row = blockIdx.x * blockDim.y + threadIdx.y;
    int lane = threadIdx.x;
    if (row >= rows) return;
    float s = 0.0f;
    for (int h = lane; h < H_; h += 32) s += (float)X[(long long)row * H_ + h] * w[h];
    for (int o = 16; o > 0; o >>= 1) s += __shfl_xor(s, o, 32);
    if (lane == 0) out[row] = s;
}

// ---------------------------------------------------------------------------
// Masked trilinear softmax over last axis (one wave per row), writes f16 aw.
// logit = qm&km ? dot + qlog[q] + klog[k] : NEG ; aw = mask ? softmax : 0
// ---------------------------------------------------------------------------
__global__ void softmax_row(const float* __restrict__ dot, int ldd,
                            const float* __restrict__ qlog, int qstride,
                            const float* __restrict__ qm, int qmstride,
                            const float* __restrict__ klog, int kstride,
                            const float* __restrict__ km, int kmstride,
                            h16* __restrict__ out, int N, int rowsPerBatch)
{
    int row  = blockIdx.x * blockDim.y + threadIdx.y;
    int lane = threadIdx.x;
    int b = row / rowsPerBatch, q = row % rowsPerBatch;
    h16* orow = out + (long long)row * N;
    if (qm[b * qmstride + q] == 0.0f) {
        for (int k = lane; k < N; k += 32) orow[k] = (h16)0.0f;
        return;
    }
    float qv = qlog[b * qstride + q];
    float ev[16];                                  // N <= 512 => <=16 per lane
    float mx = -3.4e38f;
    int nk = 0;
    for (int k = lane; k < N; k += 32) {
        float kmv = km[b * kmstride + k];
        float l = (kmv != 0.0f) ? dot[(long long)row * ldd + k] + qv + klog[b * kstride + k]
                                : NEGV;
        ev[nk++] = l;
        mx = fmaxf(mx, l);
    }
    for (int o = 16; o > 0; o >>= 1) mx = fmaxf(mx, __shfl_xor(mx, o, 32));
    float sum = 0.0f; nk = 0;
    for (int k = lane; k < N; k += 32) {
        float kmv = km[b * kmstride + k];
        float e = (kmv != 0.0f) ? __expf(ev[nk] - mx) : 0.0f;
        ev[nk++] = e; sum += e;
    }
    for (int o = 16; o > 0; o >>= 1) sum += __shfl_xor(sum, o, 32);
    float inv = (sum > 0.0f) ? 1.0f / sum : 0.0f;
    nk = 0;
    for (int k = lane; k < N; k += 32) orow[k] = (h16)(ev[nk++] * inv);
}

// Column softmax (axis=q) for kq: one thread per (b,k) column.
__global__ void softmax_col(const float* __restrict__ dot, int ldd, int M,
                            const float* __restrict__ qlog, int qstride,
                            const float* __restrict__ qm, int qmstride,
                            const float* __restrict__ klog, int kstride,
                            const float* __restrict__ km, int kmstride,
                            h16* __restrict__ out, int N)
{
    long long c = (long long)blockIdx.x * blockDim.x + threadIdx.x;  // < BL*N
    int b = (int)(c / N), k = (int)(c % N);
    if (km[b * kmstride + k] == 0.0f) {
        for (int q = 0; q < M; ++q) out[((long long)b * M + q) * N + k] = (h16)0.0f;
        return;
    }
    float kv = klog[b * kstride + k];
    float mx = -3.4e38f;
    for (int q = 0; q < M; ++q) {
        float l = (qm[b * qmstride + q] != 0.0f)
                  ? dot[((long long)b * M + q) * ldd + k] + qlog[b * qstride + q] + kv : NEGV;
        mx = fmaxf(mx, l);
    }
    float sum = 0.0f;
    for (int q = 0; q < M; ++q) {
        int v = (qm[b * qmstride + q] != 0.0f);
        float l = v ? dot[((long long)b * M + q) * ldd + k] + qlog[b * qstride + q] + kv : NEGV;
        sum += v ? __expf(l - mx) : 0.0f;
    }
    float inv = (sum > 0.0f) ? 1.0f / sum : 0.0f;
    for (int q = 0; q < M; ++q) {
        int v = (qm[b * qmstride + q] != 0.0f);
        float l = v ? dot[((long long)b * M + q) * ldd + k] + qlog[b * qstride + q] + kv : NEGV;
        out[((long long)b * M + q) * N + k] = (h16)(v ? __expf(l - mx) * inv : 0.0f);
    }
}

// ---------------------------------------------------------------------------
// Elementwise glue kernels
// ---------------------------------------------------------------------------
__global__ void cvt_f32_to_f16(const float* __restrict__ src, h16* __restrict__ dst, long long n)
{
    long long e = (long long)blockIdx.x * blockDim.x + threadIdx.x;
    if (e < n) dst[e] = (h16)src[e];
}

__global__ void feats_copy_cur(const h16* __restrict__ opt_h, int i, h16* __restrict__ feats)
{
    long long e = (long long)blockIdx.x * blockDim.x + threadIdx.x;  // BSZ*LO*H
    int h = (int)(e % H_);
    long long bt = e / H_;
    int t = (int)(bt % LO_), b = (int)(bt / LO_);
    feats[((long long)b * LO_ + t) * 7 * H_ + h] =
        opt_h[(((long long)b * NL_ + i) * LO_ + t) * (long long)H_ + h];
}

__global__ void feats_pair(const float* __restrict__ attn, const float* __restrict__ opt_f32,
                           int i, int jj, h16* __restrict__ feats)
{
    long long e = (long long)blockIdx.x * blockDim.x + threadIdx.x;  // BSZ*LO*H
    int h = (int)(e % H_);
    long long bt = e / H_;
    int t = (int)(bt % LO_), b = (int)(bt / LO_);
    float a = attn[e];
    float c = opt_f32[(((long long)b * NL_ + i) * LO_ + t) * (long long)H_ + h];
    long long base = ((long long)b * LO_ + t) * 7 * H_;
    feats[base + (1 + 2 * jj) * H_ + h] = (h16)(c * a);
    feats[base + (2 + 2 * jj) * H_ + h] = (h16)(c - a);
}

__global__ void corr_finish(const float* __restrict__ raw, const float* __restrict__ bias,
                            int i, float* __restrict__ corr_f32, h16* __restrict__ corr_h)
{
    long long e = (long long)blockIdx.x * blockDim.x + threadIdx.x;  // BSZ*LO*H
    int h = (int)(e % H_);
    long long bt = e / H_;
    int t = (int)(bt % LO_), b = (int)(bt / LO_);
    float v = tanhf(raw[e] + bias[h]);
    long long o = (((long long)b * NL_ + i) * LO_ + t) * (long long)H_ + h;
    corr_f32[o] = v;
    corr_h[o]   = (h16)v;
}

__global__ void gate_combine(const float* __restrict__ raw, const float* __restrict__ gb,
                             const float* __restrict__ opt_f32, const float* __restrict__ corr_f32,
                             h16* __restrict__ option_h, h16* __restrict__ optionT_h)
{
    int row = blockIdx.x;               // BL*LO
    int b = row / LO_, t = row % LO_;
    for (int h = threadIdx.x; h < H_; h += blockDim.x) {
        long long e = (long long)row * H_ + h;
        float g = 1.0f / (1.0f + __expf(-(raw[e] + gb[h])));
        float v = opt_f32[e] * g + corr_f32[e] * (1.0f - g);
        option_h[e] = (h16)v;
        optionT_h[((long long)b * H_ + h) * LO_ + t] = (h16)v;
    }
}

__global__ void fusion_finish(const float* __restrict__ raw, const float* __restrict__ bias,
                              const float* __restrict__ w3, float* __restrict__ fus,
                              h16* __restrict__ fus_h, h16* __restrict__ fusT_h,
                              h16* __restrict__ fus3_h)
{
    int row = blockIdx.x;               // BL*LO
    int b = row / LO_, t = row % LO_;
    for (int h = threadIdx.x; h < H_; h += blockDim.x) {
        long long e = (long long)row * H_ + h;
        float v = tanhf(raw[e] + bias[h]);
        fus[e]    = v;
        fus_h[e]  = (h16)v;
        fus3_h[e] = (h16)(v * w3[h]);
        fusT_h[((long long)b * H_ + h) * LO_ + t] = (h16)v;
    }
}

__global__ void self_prep(const float* __restrict__ f, const float* __restrict__ a2,
                          h16* __restrict__ a2_h, h16* __restrict__ prod_h,
                          h16* __restrict__ diff_h)
{
    long long e = (long long)blockIdx.x * blockDim.x + threadIdx.x;  // BL*LO*H
    float fv = f[e], av = a2[e];
    a2_h[e]   = (h16)av;
    prod_h[e] = (h16)(fv * av);
    diff_h[e] = (h16)(fv - av);
}

__global__ void final_max(const float* __restrict__ raw, const float* __restrict__ bias,
                          const float* __restrict__ opt_mask, float* __restrict__ out)
{
    int b = blockIdx.y;
    int h = blockIdx.x * blockDim.x + threadIdx.x;
    float bv = bias[h];
    float m = NEGV;
    for (int t = 0; t < LO_; ++t) {
        if (opt_mask[b * LO_ + t] != 0.0f) {
            float v = raw[(((long long)b * LO_) + t) * H_ + h] + bv;
            v = v > 0.0f ? v : 0.0f;        // relu, then masked max (masked rows = NEG)
            m = fmaxf(m, v);
        }
    }
    out[(long long)b * H_ + h] = m;
}

// ---------------------------------------------------------------------------
// Host orchestration
// ---------------------------------------------------------------------------
extern "C" void kernel_launch(void* const* d_in, const int* in_sizes, int n_in,
                              void* d_out, int out_size, void* d_ws, size_t ws_size,
                              hipStream_t stream)
{
    (void)in_sizes; (void)n_in; (void)out_size; (void)ws_size;
    const float*     ll = (const float*)d_in[0];
    const int*       am = (const int*)d_in[1];
    const long long* fp = (const long long*)d_in[2];
    const float* attn_w1 = (const float*)d_in[4];
    const float* attn_w2 = (const float*)d_in[5];
    const float* attn_w3 = (const float*)d_in[6];
    const float* opt_w1  = (const float*)d_in[7];
    const float* opt_w2  = (const float*)d_in[8];
    const float* opt_w3  = (const float*)d_in[9];
    const float* self_w1 = (const float*)d_in[10];
    const float* self_w2 = (const float*)d_in[11];
    const float* self_w3 = (const float*)d_in[12];
    const float* attn_fc_w = (const float*)d_in[13];
    const float* attn_fc_b = (const float*)d_in[14];
    const float* comp_fc_w = (const float*)d_in[15];
    const float* comp_fc_b = (const float*)d_in[16];
    const float* gate_fc_w = (const float*)d_in[17];
    const float* gate_fc_b = (const float*)d_in[18];
    const float* self_fc_w = (const float*)d_in[19];
    const float* self_fc_b = (const float*)d_in[20];
    float* outp = (float*)d_out;

    size_t off = 0;
    auto alloc = [&](size_t bytes) -> void* {
        void* p = (char*)d_ws + off;
        off += (bytes + 255) & ~(size_t)255;
        return p;
    };

    const size_t OPT = (size_t)BL_ * LO_ * H_;
    const size_t DOC = (size_t)BL_ * LD_ * H_;
    float* opt_f32  = (float*)alloc(OPT * 4);
    h16*   opt_h    = (h16*)alloc(OPT * 2);
    h16*   opt3_h   = (h16*)alloc(OPT * 2);
    h16*   optT_h   = (h16*)alloc(OPT * 2);
    h16*   doc_h    = (h16*)alloc(DOC * 2);
    h16*   doc3_h   = (h16*)alloc(DOC * 2);
    h16*   docT_h   = (h16*)alloc(DOC * 2);
    float* opt_mask = (float*)alloc((size_t)BL_ * LO_ * 4);
    float* doc_mask = (float*)alloc((size_t)BL_ * LD_ * 4);
    float* olog1    = (float*)alloc((size_t)BL_ * LO_ * 4);
    float* olog2    = (float*)alloc((size_t)BL_ * LO_ * 4);
    float* qlog3    = (float*)alloc((size_t)BL_ * LO_ * 4);
    float* slog1    = (float*)alloc((size_t)BL_ * LO_ * 4);
    float* slog2    = (float*)alloc((size_t)BL_ * LO_ * 4);
    float* dlog2    = (float*)alloc((size_t)BL_ * LD_ * 4);
    h16* comp_w_h = (h16*)alloc((size_t)H_ * 7 * H_ * 2);
    h16* attnw_h  = (h16*)alloc((size_t)H_ * 3 * H_ * 2);
    h16* gatew_h  = (h16*)alloc((size_t)H_ * 2 * H_ * 2);
    h16* selfw_h  = (h16*)alloc((size_t)H_ * 4 * H_ * 2);
    float* dotbuf = (float*)alloc((size_t)BL_ * LO_ * LD_ * 4);
    h16*  aw_h    = (h16*)alloc((size_t)BL_ * LO_ * LD_ * 2);
    h16*  kq_h    = (h16*)alloc((size_t)BL_ * LO_ * LD_ * 2);
    float* attn_f32 = (float*)alloc(OPT * 4);
    h16*   attn_h   = (h16*)alloc(OPT * 2);
    h16*   feats_h  = (h16*)alloc((size_t)BSZ_ * LO_ * 7 * H_ * 2);
    float* corr_f32 = (float*)alloc(OPT * 4);
    h16*   corr_h   = (h16*)alloc(OPT * 2);
    float* gemmout  = (float*)alloc(OPT * 4);
    h16*   option_h  = (h16*)alloc(OPT * 2);
    h16*   optionT_h = (h16*)alloc(OPT * 2);
    float* cow_f32 = (float*)alloc((size_t)BL_ * LO_ * LO_ * 4);
    h16*   cow_h   = (h16*)alloc((size_t)BL_ * LO_ * LO_ * 2);
    float* fus_f32 = (float*)alloc(OPT * 4);
    h16*   fus_h   = (h16*)alloc(OPT * 2);
    h16*   fusT_h  = (h16*)alloc(OPT * 2);
    h16*   fus3_h  = (h16*)alloc(OPT * 2);
    h16*   prod_h  = (h16*)alloc(OPT * 2);
    h16*   diff_h  = (h16*)alloc(OPT * 2);

    auto gemm = [&](const h16* A, long long sA, int lda, const h16* B, long long sB, int ldb,
                    float* C, long long sC, int ldc, int M, int N, int K, int nb, int accum) {
        long long tiles = (long long)nb * (M / 32) * (N / 64);
        int blocks = (int)((tiles + 7) / 8);
        wmma_gemm_nt<<<blocks, dim3(32, 8), 0, stream>>>(A, sA, lda, B, sB, ldb,
                                                         C, sC, ldc, M, N, K, nb, accum);
    };
    auto cvt = [&](const float* s, h16* dst, long long n) {
        cvt_f32_to_f16<<<(int)((n + 255) / 256), 256, 0, stream>>>(s, dst, n);
    };

    // Phase 0: f16 weights
    cvt(comp_fc_w, comp_w_h, (long long)H_ * 7 * H_);
    cvt(attn_fc_w, attnw_h,  (long long)H_ * 3 * H_);
    cvt(gate_fc_w, gatew_h,  (long long)H_ * 2 * H_);
    cvt(self_fc_w, selfw_h,  (long long)H_ * 4 * H_);

    // Phase 1: gather + per-row logits
    prep_opt<<<BL_ * LO_, 256, 0, stream>>>(ll, am, fp, opt_w3, opt_f32, opt_h, opt3_h, optT_h, opt_mask);
    prep_doc<<<BL_ * LD_, 256, 0, stream>>>(ll, am, fp, attn_w3, doc_h, doc3_h, docT_h, doc_mask);
    rowdot_f32<<<BL_ * LO_ / 8, dim3(32, 8), 0, stream>>>(opt_f32, opt_w1, olog1, BL_ * LO_);
    rowdot_f32<<<BL_ * LO_ / 8, dim3(32, 8), 0, stream>>>(opt_f32, opt_w2, olog2, BL_ * LO_);
    rowdot_f16<<<BL_ * LD_ / 8, dim3(32, 8), 0, stream>>>(doc_h, attn_w2, dlog2, BL_ * LD_);

    // Phase 2: option-vs-option correlation + comp_fc
    const long long sOPT = (long long)NL_ * LO_ * H_;
    for (int i = 0; i < NL_; ++i) {
        feats_copy_cur<<<(BSZ_ * LO_ * H_) / 256, 256, 0, stream>>>(opt_h, i, feats_h);
        int jj = 0;
        for (int j = 0; j < NL_; ++j) {
            if (j == i) continue;
            gemm(opt_h + (long long)i * LO_ * H_, sOPT, H_,
                 opt3_h + (long long)j * LO_ * H_, sOPT, H_,
                 dotbuf, (long long)LO_ * LO_, LO_, LO_, LO_, H_, BSZ_, 0);
            softmax_row<<<(BSZ_ * LO_) / 4, dim3(32, 4), 0, stream>>>(
                 dotbuf, LO_,
                 olog1 + i * LO_, NL_ * LO_, opt_mask + i * LO_, NL_ * LO_,
                 olog2 + j * LO_, NL_ * LO_, opt_mask + j * LO_, NL_ * LO_,
                 aw_h, LO_, LO_);
            gemm(aw_h, (long long)LO_ * LO_, LO_,
                 optT_h + (long long)j * H_ * LO_, (long long)NL_ * H_ * LO_, LO_,
                 attn_f32, (long long)LO_ * H_, H_, LO_, H_, LO_, BSZ_, 0);
            feats_pair<<<(BSZ_ * LO_ * H_) / 256, 256, 0, stream>>>(attn_f32, opt_f32, i, jj, feats_h);
            ++jj;
        }
        gemm(feats_h, (long long)LO_ * 7 * H_, 7 * H_, comp_w_h, 0, 7 * H_,
             gemmout, (long long)LO_ * H_, H_, LO_, H_, 7 * H_, BSZ_, 0);
        corr_finish<<<(BSZ_ * LO_ * H_) / 256, 256, 0, stream>>>(gemmout, comp_fc_b, i, corr_f32, corr_h);
    }

    // Phase 3: gate + option
    gemm(opt_h,  (long long)LO_ * H_, H_, gatew_h,      0, 2 * H_,
         gemmout, (long long)LO_ * H_, H_, LO_, H_, H_, BL_, 0);
    gemm(corr_h, (long long)LO_ * H_, H_, gatew_h + H_, 0, 2 * H_,
         gemmout, (long long)LO_ * H_, H_, LO_, H_, H_, BL_, 1);
    gate_combine<<<BL_ * LO_, 256, 0, stream>>>(gemmout, gate_fc_b, opt_f32, corr_f32,
                                                option_h, optionT_h);
    rowdot_f16<<<BL_ * LO_ / 8, dim3(32, 8), 0, stream>>>(option_h, attn_w1, qlog3, BL_ * LO_);

    // Phase 4: option <-> doc attention + attn_fc
    gemm(option_h, (long long)LO_ * H_, H_, doc3_h, (long long)LD_ * H_, H_,
         dotbuf, (long long)LO_ * LD_, LD_, LO_, LD_, H_, BL_, 0);
    softmax_row<<<(BL_ * LO_) / 4, dim3(32, 4), 0, stream>>>(
         dotbuf, LD_, qlog3, LO_, opt_mask, LO_, dlog2, LD_, doc_mask, LD_, aw_h, LD_, LO_);
    softmax_col<<<(BL_ * LD_) / 256, 256, 0, stream>>>(
         dotbuf, LD_, LO_, qlog3, LO_, opt_mask, LO_, dlog2, LD_, doc_mask, LD_, kq_h, LD_);
    gemm(aw_h, (long long)LO_ * LD_, LD_, docT_h, (long long)H_ * LD_, LD_,
         attn_f32, (long long)LO_ * H_, H_, LO_, H_, LD_, BL_, 0);
    gemm(aw_h, (long long)LO_ * LD_, LD_, kq_h, (long long)LO_ * LD_, LD_,
         cow_f32, (long long)LO_ * LO_, LO_, LO_, LO_, LD_, BL_, 0);
    cvt(cow_f32, cow_h, (long long)BL_ * LO_ * LO_);
    float* coat_f32 = corr_f32;   // corr dead after gate -> reuse
    h16*   coat_h   = corr_h;
    gemm(cow_h, (long long)LO_ * LO_, LO_, optionT_h, (long long)H_ * LO_, LO_,
         coat_f32, (long long)LO_ * H_, H_, LO_, H_, LO_, BL_, 0);
    cvt(attn_f32, attn_h, (long long)OPT);
    cvt(coat_f32, coat_h, (long long)OPT);
    gemm(option_h, (long long)LO_ * H_, H_, attnw_h,          0, 3 * H_,
         gemmout, (long long)LO_ * H_, H_, LO_, H_, H_, BL_, 0);
    gemm(attn_h,   (long long)LO_ * H_, H_, attnw_h + H_,     0, 3 * H_,
         gemmout, (long long)LO_ * H_, H_, LO_, H_, H_, BL_, 1);
    gemm(coat_h,   (long long)LO_ * H_, H_, attnw_h + 2 * H_, 0, 3 * H_,
         gemmout, (long long)LO_ * H_, H_, LO_, H_, H_, BL_, 1);
    fusion_finish<<<BL_ * LO_, 256, 0, stream>>>(gemmout, attn_fc_b, self_w3,
                                                 fus_f32, fus_h, fusT_h, fus3_h);
    rowdot_f32<<<BL_ * LO_ / 8, dim3(32, 8), 0, stream>>>(fus_f32, self_w1, slog1, BL_ * LO_);
    rowdot_f32<<<BL_ * LO_ / 8, dim3(32, 8), 0, stream>>>(fus_f32, self_w2, slog2, BL_ * LO_);

    // Phase 5: self attention + self_fc + masked max
    gemm(fus_h, (long long)LO_ * H_, H_, fus3_h, (long long)LO_ * H_, H_,
         dotbuf, (long long)LO_ * LO_, LO_, LO_, LO_, H_, BL_, 0);
    softmax_row<<<(BL_ * LO_) / 4, dim3(32, 4), 0, stream>>>(
         dotbuf, LO_, slog1, LO_, opt_mask, LO_, slog2, LO_, opt_mask, LO_, aw_h, LO_, LO_);
    float* attn2_f32 = attn_f32;  // reuse
    h16*   a2_h      = attn_h;
    gemm(aw_h, (long long)LO_ * LO_, LO_, fusT_h, (long long)H_ * LO_, LO_,
         attn2_f32, (long long)LO_ * H_, H_, LO_, H_, LO_, BL_, 0);
    self_prep<<<(BL_ * LO_ * H_) / 256, 256, 0, stream>>>(fus_f32, attn2_f32, a2_h, prod_h, diff_h);
    gemm(fus_h,  (long long)LO_ * H_, H_, selfw_h,          0, 4 * H_,
         gemmout, (long long)LO_ * H_, H_, LO_, H_, H_, BL_, 0);
    gemm(a2_h,   (long long)LO_ * H_, H_, selfw_h + H_,     0, 4 * H_,
         gemmout, (long long)LO_ * H_, H_, LO_, H_, H_, BL_, 1);
    gemm(prod_h, (long long)LO_ * H_, H_, selfw_h + 2 * H_, 0, 4 * H_,
         gemmout, (long long)LO_ * H_, H_, LO_, H_, H_, BL_, 1);
    gemm(diff_h, (long long)LO_ * H_, H_, selfw_h + 3 * H_, 0, 4 * H_,
         gemmout, (long long)LO_ * H_, H_, LO_, H_, H_, BL_, 1);
    final_max<<<dim3(H_ / 256, BL_), 256, 0, stream>>>(gemmout, self_fc_b, opt_mask, outp);
}
